// MultiHeadSelfAttention_58540404244815
// MI455X (gfx1250) — matrix operand order repaired and checked by reading, not compiled, not run on hardware
//
#include <hip/hip_runtime.h>
#include <hip/hip_bf16.h>

typedef __attribute__((ext_vector_type(16))) _Float16 v16h;
typedef __attribute__((ext_vector_type(8)))  _Float16 v8h;
typedef __attribute__((ext_vector_type(8)))  float    v8f;

#define SEQ    2048
#define DMODEL 2048
#define NH     16
#define HD     128
#define D3     6144

// ---------------------------------------------------------------------------
// Async global->LDS copy path (CDNA5): global_load_async_to_lds_b128 tracked
// by ASYNCcnt, waited with s_wait_asynccnt. Builtin signature (from clang
// diagnostic): (v4i addrspace(1)*, v4i addrspace(3)*, imm offset, imm cpol).
#if defined(__gfx1250__) && \
    __has_builtin(__builtin_amdgcn_global_load_async_to_lds_b128) && \
    __has_builtin(__builtin_amdgcn_s_wait_asynccnt)
#define ASYNC_LDS 1
#endif

#ifdef ASYNC_LDS
typedef int v4i_vs __attribute__((vector_size(16)));
typedef __attribute__((address_space(1))) v4i_vs* gas_ptr;
typedef __attribute__((address_space(3))) v4i_vs* las_ptr;
#endif

// Copy 16 bytes from global to LDS (per lane).
__device__ __forceinline__ void copy16_g2l(const void* g, void* l) {
#ifdef ASYNC_LDS
    __builtin_amdgcn_global_load_async_to_lds_b128((gas_ptr)g, (las_ptr)l, 0, 0);
#else
    *(v8h*)l = *(const v8h*)g;
#endif
}
__device__ __forceinline__ void copy_wait() {
#ifdef ASYNC_LDS
    __builtin_amdgcn_s_wait_asynccnt(0);
#endif
}

// ---------------------------------------------------------------------------
// WMMA fragment loaders (wave32, v_wmma_f32_16x16x32_f16)
//
// A (16x32 f16, MxK): lane L holds row m = L&15;
//   halves 0..7  -> K = (L>>4)*8 + i ; halves 8..15 -> +16.
__device__ __forceinline__ v16h load_A_frag(const _Float16* base, int ld,
                                            int row_off, int kb, int lane) {
    const int m = lane & 15, g = lane >> 4;
    const _Float16* p = base + (size_t)(row_off + m) * ld + kb + g * 8;
    v8h lo = *(const v8h*)p;
    v8h hi = *(const v8h*)(p + 16);
    v16h r;
#pragma unroll
    for (int i = 0; i < 8; ++i) { r[i] = lo[i]; r[8 + i] = hi[i]; }
    return r;
}

// B (32x16 f16, KxN): lane L holds column n = L&15; halves 0..7 -> K=(L>>4)*8+i; 8..15 -> +16.
// Two 16-byte pieces (k, k+16) of "row n" of a row-major [N, ld] matrix (B = W^T).
__device__ __forceinline__ v16h load_B_frag(const _Float16* base, int ld,
                                            int col_off, int kb, int lane) {
    const int n = lane & 15, g = lane >> 4;
    const _Float16* p = base + (size_t)(col_off + n) * ld + kb + g * 8;
    v8h lo = *(const v8h*)p;
    v8h hi = *(const v8h*)(p + 16);
    v16h r;
#pragma unroll
    for (int i = 0; i < 8; ++i) { r[i] = lo[i]; r[8 + i] = hi[i]; }
    return r;
}

// ---------------------------------------------------------------------------
__global__ void cvt_f32_f16(const float* __restrict__ in,
                            _Float16* __restrict__ out, int n) {
    int i = blockIdx.x * blockDim.x + threadIdx.x;
    if (i < n) out[i] = (_Float16)in[i];
}

// ---------------------------------------------------------------------------
// C[M,N] = A[M,K] * B[N,K]^T. 256 threads = 8 waves; block tile 128x128.
// Shared B tile (128 cols x 32 k) double-buffered in LDS via async copies.
#define BSTR 40   // padded LDS row stride in halves (32 data + 8 pad)
template <bool HALF_OUT>
__global__ __launch_bounds__(256)
void gemm_nt(const _Float16* __restrict__ A, const _Float16* __restrict__ B,
             void* __restrict__ Cv, int M, int N, int K) {
    __shared__ _Float16 ldsB[2][128 * BSTR];

    const int lane = threadIdx.x & 31;
    const int wave = threadIdx.x >> 5;
    const int tid  = threadIdx.x;
    const int row0 = (blockIdx.x * 8 + wave) * 16;
    const int col0 = blockIdx.y * 128;

    // Stage 128x32 B tile: 512 chunks of 16B; each thread copies 2 chunks.
    auto stageB = [&](int buf, int k0) {
#pragma unroll
        for (int c2 = 0; c2 < 2; ++c2) {
            const int c   = tid * 2 + c2;
            const int row = c >> 2;         // 4 chunks (64B) per row
            const int off = (c & 3) * 8;    // halves within row
            copy16_g2l(B + (size_t)(col0 + row) * K + k0 + off,
                       &ldsB[buf][row * BSTR + off]);
        }
    };

    stageB(0, 0);
    v8f acc[8] = {};
    int buf = 0;
    for (int k0 = 0; k0 < K; k0 += 32) {
        copy_wait();
        __syncthreads();                       // buf ready; prev reads done
        if (k0 + 32 < K) stageB(buf ^ 1, k0 + 32);
        v16h a = load_A_frag(A, K, row0, k0, lane);
#pragma unroll
        for (int t = 0; t < 8; ++t) {
            v16h b = load_B_frag(&ldsB[buf][0], BSTR, t * 16, 0, lane);
            acc[t] = __builtin_amdgcn_wmma_f32_16x16x32_f16(
                false, a, false, b, (short)0, acc[t], false, false);
        }
        buf ^= 1;
    }

    const int n = lane & 15, g = lane >> 4;
#pragma unroll
    for (int t = 0; t < 8; ++t) {
#pragma unroll
        for (int r = 0; r < 8; ++r) {
            const size_t idx = (size_t)(row0 + r + 8 * g) * N + col0 + t * 16 + n;
            if (HALF_OUT) ((_Float16*)Cv)[idx] = (_Float16)acc[t][r];
            else          ((float*)Cv)[idx]    = acc[t][r];
        }
    }
}

// ---------------------------------------------------------------------------
// RoPE (rotate-half) + rearrange into head-major Q,K [H,S,hd] and V^T [H,hd,S]
__global__ void rope_rearrange(const _Float16* __restrict__ qkv,
                               _Float16* __restrict__ qh,
                               _Float16* __restrict__ kh,
                               _Float16* __restrict__ vT) {
    const int d = threadIdx.x;   // 0..127
    const int s = blockIdx.x;    // 0..2047
    const int h = blockIdx.y;    // 0..15
    const size_t rbase = (size_t)s * D3 + h * HD;

    float qv = (float)qkv[rbase + d];
    float kv = (float)qkv[rbase + DMODEL + d];
    float vv = (float)qkv[rbase + 2 * DMODEL + d];

    const int dp = (d < 64) ? d + 64 : d - 64;
    float qp = (float)qkv[rbase + dp];
    float kp = (float)qkv[rbase + DMODEL + dp];

    const int j = d & 63;
    float inv = __powf(10000.0f, -(float)(2 * j) / 128.0f);
    float ang = (float)s * inv;
    float sn, c;
    __sincosf(ang, &sn, &c);
    const float sgn = (d < 64) ? -1.0f : 1.0f;

    const size_t hs = ((size_t)h * SEQ + s) * HD + d;
    qh[hs] = (_Float16)(qv * c + sgn * qp * sn);
    kh[hs] = (_Float16)(kv * c + sgn * kp * sn);
    vT[((size_t)h * HD + d) * SEQ + s] = (_Float16)vv;
}

// ---------------------------------------------------------------------------
// Flash attention: block = (q-tile of 128, head); 8 waves, 16 queries/wave.
// K tile (32 keys x 128 d) and V^T tile (128 d x 32 keys) are shared by all
// waves -> double-buffered async LDS staging.
#define KSTR 136   // K tile LDS row stride in halves (128 + 8 pad)
#define VSTR 40    // V tile LDS row stride in halves (32 + 8 pad)
__global__ __launch_bounds__(256)
void attn_fwd(const _Float16* __restrict__ qh, const _Float16* __restrict__ kh,
              const _Float16* __restrict__ vT, _Float16* __restrict__ att) {
    __shared__ _Float16 ldsK[2][32 * KSTR];
    __shared__ _Float16 ldsV[2][128 * VSTR];
    __shared__ _Float16 ldsP[8][16 * 32];   // per-wave P tile (16 q x 32 k)

    const int lane = threadIdx.x & 31;
    const int wave = threadIdx.x >> 5;
    const int tid  = threadIdx.x;
    const int h    = blockIdx.y;
    const int q0   = blockIdx.x * 128 + wave * 16;

    const _Float16* Qh = qh + (size_t)h * SEQ * HD;
    const _Float16* Kh = kh + (size_t)h * SEQ * HD;
    const _Float16* Vh = vT + (size_t)h * HD * SEQ;

    // Stage 32x128 K tile: 512 chunks of 16B (16 per row of 256B).
    auto stageK = [&](int buf, int kt) {
#pragma unroll
        for (int c2 = 0; c2 < 2; ++c2) {
            const int c   = tid * 2 + c2;
            const int row = c >> 4;
            const int off = (c & 15) * 8;
            copy16_g2l(Kh + (size_t)(kt + row) * HD + off,
                       &ldsK[buf][row * KSTR + off]);
        }
    };
    // Stage 128x32 V^T tile: 512 chunks of 16B (4 per row of 64B).
    auto stageV = [&](int buf, int kt) {
#pragma unroll
        for (int c2 = 0; c2 < 2; ++c2) {
            const int c   = tid * 2 + c2;
            const int row = c >> 2;
            const int off = (c & 3) * 8;
            copy16_g2l(Vh + (size_t)row * SEQ + kt + off,
                       &ldsV[buf][row * VSTR + off]);
        }
    };

    v16h aQ[4];
#pragma unroll
    for (int j = 0; j < 4; ++j) aQ[j] = load_A_frag(Qh, HD, q0, 32 * j, lane);

    v8f o[8] = {};
    float m_i[8], l_i[8];
#pragma unroll
    for (int r = 0; r < 8; ++r) { m_i[r] = -3.0e38f; l_i[r] = 0.0f; }

    const float scale = 0.08838834764831845f;  // 1/sqrt(128)
    const int n = lane & 15, g = lane >> 4;

    stageK(0, 0);
    stageV(0, 0);
    int buf = 0;
    for (int kt = 0; kt < SEQ; kt += 32) {
        copy_wait();
        __syncthreads();
        if (kt + 32 < SEQ) { stageK(buf ^ 1, kt + 32); stageV(buf ^ 1, kt + 32); }

        // scores: Q(16x128) x K^T -> two 16x16 f32 fragments (from LDS)
        v8f sc0 = {}, sc1 = {};
#pragma unroll
        for (int j = 0; j < 4; ++j) {
            v16h b0 = load_B_frag(&ldsK[buf][0], KSTR, 0, 32 * j, lane);
            sc0 = __builtin_amdgcn_wmma_f32_16x16x32_f16(
                false, aQ[j], false, b0, (short)0, sc0, false, false);
            v16h b1 = load_B_frag(&ldsK[buf][0], KSTR, 16, 32 * j, lane);
            sc1 = __builtin_amdgcn_wmma_f32_16x16x32_f16(
                false, aQ[j], false, b1, (short)0, sc1, false, false);
        }

        // online softmax (row = r + 8*g, reductions across 16-lane groups)
        float alpha[8];
#pragma unroll
        for (int r = 0; r < 8; ++r) {
            float v = fmaxf(sc0[r], sc1[r]);
#pragma unroll
            for (int off = 1; off < 16; off <<= 1)
                v = fmaxf(v, __shfl_xor(v, off, 16));
            const float mn = fmaxf(m_i[r], v * scale);
            alpha[r] = __expf(m_i[r] - mn);
            m_i[r] = mn;
            const float p0 = __expf(sc0[r] * scale - mn);
            const float p1 = __expf(sc1[r] * scale - mn);
            ldsP[wave][(r + 8 * g) * 32 + n]      = (_Float16)p0;
            ldsP[wave][(r + 8 * g) * 32 + 16 + n] = (_Float16)p1;
            float srow = p0 + p1;
#pragma unroll
            for (int off = 1; off < 16; off <<= 1)
                srow += __shfl_xor(srow, off, 16);
            l_i[r] = l_i[r] * alpha[r] + srow;
        }
#pragma unroll
        for (int t = 0; t < 8; ++t)
#pragma unroll
            for (int r = 0; r < 8; ++r) o[t][r] *= alpha[r];

        // O += P(16x32) x V(32x128); ldsP converts C-layout -> A-layout
        v16h pA = load_A_frag(&ldsP[wave][0], 32, 0, 0, lane);
#pragma unroll
        for (int t = 0; t < 8; ++t) {
            v16h bV = load_B_frag(&ldsV[buf][0], VSTR, t * 16, 0, lane);
            o[t] = __builtin_amdgcn_wmma_f32_16x16x32_f16(
                false, pA, false, bV, (short)0, o[t], false, false);
        }
        buf ^= 1;
    }

#pragma unroll
    for (int t = 0; t < 8; ++t) {
#pragma unroll
        for (int r = 0; r < 8; ++r) {
            const size_t row = q0 + r + 8 * g;
            att[row * DMODEL + h * HD + t * 16 + n] =
                (_Float16)(o[t][r] / l_i[r]);
        }
    }
}

// ---------------------------------------------------------------------------
extern "C" void kernel_launch(void* const* d_in, const int* in_sizes, int n_in,
                              void* d_out, int out_size, void* d_ws, size_t ws_size,
                              hipStream_t stream) {
    const float* x     = (const float*)d_in[0];
    const float* w_qkv = (const float*)d_in[1];
    const float* w_out = (const float*)d_in[2];
    float* out = (float*)d_out;

    char* ws = (char*)d_ws;
    _Float16* x_h    = (_Float16*)ws; ws += (size_t)SEQ * DMODEL * 2;
    _Float16* wqkv_h = (_Float16*)ws; ws += (size_t)D3 * DMODEL * 2;
    _Float16* wout_h = (_Float16*)ws; ws += (size_t)DMODEL * DMODEL * 2;
    _Float16* qkv_h  = (_Float16*)ws; ws += (size_t)SEQ * D3 * 2;
    _Float16* qh     = (_Float16*)ws; ws += (size_t)SEQ * DMODEL * 2;
    _Float16* kh     = (_Float16*)ws; ws += (size_t)SEQ * DMODEL * 2;
    _Float16* vT     = (_Float16*)ws; ws += (size_t)SEQ * DMODEL * 2;
    _Float16* att_h  = (_Float16*)ws; ws += (size_t)SEQ * DMODEL * 2;

    cvt_f32_f16<<<(SEQ * DMODEL + 255) / 256, 256, 0, stream>>>(x, x_h, SEQ * DMODEL);
    cvt_f32_f16<<<(D3 * DMODEL + 255) / 256, 256, 0, stream>>>(w_qkv, wqkv_h, D3 * DMODEL);
    cvt_f32_f16<<<(DMODEL * DMODEL + 255) / 256, 256, 0, stream>>>(w_out, wout_h, DMODEL * DMODEL);

    gemm_nt<true><<<dim3(SEQ / 128, D3 / 128), 256, 0, stream>>>(
        x_h, wqkv_h, (void*)qkv_h, SEQ, D3, DMODEL);

    rope_rearrange<<<dim3(SEQ, NH), HD, 0, stream>>>(qkv_h, qh, kh, vT);

    attn_fwd<<<dim3(SEQ / 128, NH), 256, 0, stream>>>(qh, kh, vT, att_h);

    gemm_nt<false><<<dim3(SEQ / 128, DMODEL / 128), 256, 0, stream>>>(
        att_h, wout_h, (void*)out, SEQ, DMODEL, DMODEL);
}
